// PartChannelAdaptiveConv_9861244912119
// MI455X (gfx1250) — compile-verified, matrix operand order checked by reading, not run on hardware
//
#include <hip/hip_runtime.h>

typedef _Float16 half_t;
typedef __attribute__((ext_vector_type(16))) _Float16 v16h;
typedef __attribute__((ext_vector_type(8)))  _Float16 v8h;
typedef __attribute__((ext_vector_type(8)))  float    v8f;
typedef __attribute__((ext_vector_type(4)))  unsigned uint32x4;
typedef __attribute__((ext_vector_type(8)))  int      int32x8;
typedef __attribute__((ext_vector_type(4)))  int      int32x4;

#define B_    32
#define C_    128
#define CIN_  64
#define COUT_ 128
#define H_    128
#define W_    128
#define KTOT  576      // CIN_*9
#define KSTEP 32
#define NSTEPS 18      // 576/32
#define NTILE 64
#define LROW  40       // 32 halfs + 8 pad (80B row pitch: conflict-free b128 access)

// ---------------- Kernel 1: per-(b,c) spatial sums ----------------
__global__ __launch_bounds__(256) void k_chansum(const float* __restrict__ x,
                                                 float* __restrict__ sums) {
  int bc = blockIdx.x;
  const float* p = x + (size_t)bc * (H_ * W_);
  float s = 0.f;
  for (int i = threadIdx.x; i < H_ * W_; i += 256) s += p[i];
  __shared__ float red[256];
  red[threadIdx.x] = s;
  __syncthreads();
  for (int off = 128; off > 0; off >>= 1) {
    if (threadIdx.x < off) red[threadIdx.x] += red[threadIdx.x + off];
    __syncthreads();
  }
  if (threadIdx.x == 0) sums[bc] = red[0];
}

// ---------------- Kernel 2: stable rank selection (argsort[:, :64]) ----------
__global__ __launch_bounds__(128) void k_select(const float* __restrict__ sums,
                                                int* __restrict__ sel) {
  int b = blockIdx.x;
  __shared__ float s[C_];
  int c = threadIdx.x;
  s[c] = sums[b * C_ + c];
  __syncthreads();
  float mine = s[c];
  int rank = 0;
  for (int c2 = 0; c2 < C_; ++c2) {
    float v = s[c2];
    rank += (v < mine) || (v == mine && c2 < c);   // stable ascending
  }
  if (rank < CIN_) sel[b * CIN_ + rank] = c;
}

// ---------------- Kernel 3: weights f32 -> f16 [Cout][576] ----------------
__global__ __launch_bounds__(256) void k_wconv(const float* __restrict__ w,
                                               half_t* __restrict__ wh) {
  int i = blockIdx.x * 256 + threadIdx.x;
  if (i < COUT_ * KTOT) wh[i] = (half_t)w[i];
}

// ---------------- Kernel 4: implicit-GEMM conv via WMMA f16 + TDM ----------
// Block tile: M=128 (all Cout) x N=64 (one 64-wide segment of an image row).
// 8 waves (wave32) arranged 4(M) x 2(N); each wave: 2x2 fragments of 16x16.
// A tile (weights): per-wave Tensor Data Mover DMA (2D tile 16x32 halfs,
// row stride 576 halfs), TDM padding reproduces the LROW=40 pitch.
// B tile (im2col): software-pipelined global gather; each thread owns 8
// consecutive K slots of one column -> single ds_store_b128 per step.
__global__ __launch_bounds__(256) void k_conv(const float* __restrict__ x,
                                              const half_t* __restrict__ wh,
                                              const float* __restrict__ bias,
                                              const int* __restrict__ sel,
                                              float* __restrict__ out) {
  __shared__ half_t lA[COUT_ * LROW];   // weights tile  [co][K 32 + pad]
  __shared__ half_t lB[NTILE * LROW];   // im2col tile   [n ][K 32 + pad]
  __shared__ int selC[CIN_];

  int bid = blockIdx.x;
  int wt = bid & 1;
  int h  = (bid >> 1) & (H_ - 1);
  int b  = bid >> 8;
  int w0 = wt * NTILE;

  int tid  = threadIdx.x;
  int lane = tid & 31;
  int wave = tid >> 5;
  int waveM = wave >> 1;            // co base waveM*32
  int waveN = wave & 1;             // n  base waveN*32
  int hh  = lane >> 4;
  int l15 = lane & 15;

  if (tid < CIN_) selC[tid] = sel[b * CIN_ + tid];
  __syncthreads();

  const float* xb = x + (size_t)b * C_ * H_ * W_;

  // --- TDM descriptor: constant part (group 1), per ISA 08 §8.4 ---
  // word0: data_size=1 (2B) bit16; pad_enable bit20; pad_interval=3 (16 DW)
  //        bits24:22; pad_amount=3 (4 DW) bits31:25; workgroup_mask=0.
  int32x8 g1;
  g1[0] = (1 << 16) | (1 << 20) | (3 << 22) | (3 << 25);
  g1[1] = (int)((KTOT & 0xFFFF) << 16);  // tensor_dim0[15:0] = 576
  g1[2] = (16 << 16);                    // tensor_dim1 = 16 (this wave's slice)
  g1[3] = (32 << 16);                    // tile_dim0 = 32
  g1[4] = 16;                            // tile_dim1 = 16
  g1[5] = KTOT;                          // tensor_dim0_stride = 576
  g1[6] = 0;
  g1[7] = 0;
  int32x4 gz = {0, 0, 0, 0};
#if defined(__clang_major__) && __clang_major__ >= 23
  int32x8 gz8 = {0, 0, 0, 0, 0, 0, 0, 0};
#endif
  unsigned ldsA = (unsigned)(size_t)(void*)lA + (unsigned)(wave * 16 * LROW * 2);
  unsigned long long gbase =
      (unsigned long long)(size_t)(wh + (size_t)(wave * 16) * KTOT);

  // --- im2col software pipeline: thread u -> column n = u>>2,
  //     8 consecutive K slots j = (u&3)*8 + i  (one b128 LDS store) ---
  int nB    = tid >> 2;
  int jbase = (tid & 3) * 8;
  int ciS[8], rS[8];
  float pre[8];
#pragma unroll
  for (int i = 0; i < 8; ++i) {
    int j = jbase + i;
    int ci = j / 9;                 // init-only divide (j < 32)
    int r  = j - 9 * ci;
    ciS[i] = ci; rS[i] = r;
    // prefetch t=0
    int kh = (r >= 3) + (r >= 6);
    int kw = r - 3 * kh;
    int ch = selC[ci];
    int hp = h + kh - 1;
    int wp = w0 + nB + kw - 1;
    float v = 0.f;
    if ((unsigned)hp < (unsigned)H_ && (unsigned)wp < (unsigned)W_)
      v = xb[((size_t)ch * H_ + hp) * W_ + wp];
    pre[i] = v;
  }
  half_t* lbDst = &lB[nB * LROW + jbase];   // 16B aligned (80*n + 16*(u&3))

  v8f acc[2][2] = {};

  for (int t = 0; t < NSTEPS; ++t) {
    __syncthreads();  // previous step's LDS consumers done

    // ---- A tile: wave-level TDM DMA of 16x32 halfs into padded LDS rows ----
    {
      unsigned long long ga = gbase + (unsigned long long)(t * KSTEP) * 2;
      uint32x4 g0;
      g0[0] = 1u;                                   // count=1 (valid D#)
      g0[1] = ldsA;                                 // lds_addr (bytes)
      g0[2] = (unsigned)ga;                         // global_addr[31:0]
      g0[3] = (unsigned)((ga >> 32) & 0x1FFFFFFu)   // global_addr[56:32]
              | (2u << 30);                         // type=2 ("image")
#if defined(__clang_major__) && __clang_major__ >= 23
      __builtin_amdgcn_tensor_load_to_lds(g0, g1, gz, gz, gz8, 0);
#else
      __builtin_amdgcn_tensor_load_to_lds(g0, g1, gz, gz, 0);
#endif
    }

    // ---- B tile: pack prefetched values, single b128 LDS store ----
    {
      v8h vec;
#pragma unroll
      for (int i = 0; i < 8; ++i) vec[i] = (half_t)pre[i];
      *(v8h*)lbDst = vec;
    }

    __builtin_amdgcn_s_wait_tensorcnt(0);  // this wave's A slice landed
    __syncthreads();                       // all slices + B visible

    // ---- prefetch step t+1 (overlaps with the WMMAs below) ----
    if (t + 1 < NSTEPS) {
#pragma unroll
      for (int i = 0; i < 8; ++i) {
        int r = rS[i];
        int ci = ciS[i] + 3 + (r >= 4);    // K += 32 incremental decode
        r = (r < 4) ? r + 5 : r - 4;
        ciS[i] = ci; rS[i] = r;
        int kh = (r >= 3) + (r >= 6);
        int kw = r - 3 * kh;
        int ch = selC[ci];
        int hp = h + kh - 1;
        int wp = w0 + nB + kw - 1;
        float v = 0.f;
        if ((unsigned)hp < (unsigned)H_ && (unsigned)wp < (unsigned)W_)
          v = xb[((size_t)ch * H_ + hp) * W_ + wp];
        pre[i] = v;
      }
    }

    // ---- fragment loads (ISA 16-bit operand layout) + 2x2 WMMA ----
    v16h afrag[2], bfrag[2];
#pragma unroll
    for (int mf = 0; mf < 2; ++mf) {
      int m = waveM * 32 + mf * 16 + l15;
      v8h lo = *(const v8h*)&lA[m * LROW + 8 * hh];
      v8h hi = *(const v8h*)&lA[m * LROW + 16 + 8 * hh];
#pragma unroll
      for (int q = 0; q < 8; ++q) { afrag[mf][q] = lo[q]; afrag[mf][q + 8] = hi[q]; }
    }
#pragma unroll
    for (int nf = 0; nf < 2; ++nf) {
      int n = waveN * 32 + nf * 16 + l15;
      v8h lo = *(const v8h*)&lB[n * LROW + 8 * hh];
      v8h hi = *(const v8h*)&lB[n * LROW + 16 + 8 * hh];
#pragma unroll
      for (int q = 0; q < 8; ++q) { bfrag[nf][q] = lo[q]; bfrag[nf][q + 8] = hi[q]; }
    }
#pragma unroll
    for (int mf = 0; mf < 2; ++mf)
#pragma unroll
      for (int nf = 0; nf < 2; ++nf)
        acc[mf][nf] = __builtin_amdgcn_wmma_f32_16x16x32_f16(
            false, afrag[mf], false, bfrag[nf],
            (short)0, acc[mf][nf], false, false);
  }

  // ---- epilogue: bias + f32 store (lanes 0-15 contiguous in w) ----
#pragma unroll
  for (int mf = 0; mf < 2; ++mf)
#pragma unroll
    for (int nf = 0; nf < 2; ++nf) {
      int n = w0 + waveN * 32 + nf * 16 + l15;
#pragma unroll
      for (int r = 0; r < 8; ++r) {
        int co = waveM * 32 + mf * 16 + 8 * hh + r;
        out[(((size_t)b * COUT_ + co) * H_ + h) * W_ + n] =
            acc[mf][nf][r] + bias[co];
      }
    }
}

// ---------------- Host-side launcher ----------------
extern "C" void kernel_launch(void* const* d_in, const int* in_sizes, int n_in,
                              void* d_out, int out_size, void* d_ws, size_t ws_size,
                              hipStream_t stream) {
  const float* x    = (const float*)d_in[0];   // [32,128,128,128] f32
  const float* w    = (const float*)d_in[1];   // [128,64,3,3] f32
  const float* bias = (const float*)d_in[2];   // [128] f32
  float* out = (float*)d_out;                  // [32,128,128,128] f32

  char* ws = (char*)d_ws;
  float*  sums = (float*)ws;                                   // 4096 f32
  int*    sel  = (int*)(ws + 4096 * sizeof(float));            // 2048 i32
  half_t* whf  = (half_t*)(ws + 4096 * 4 + 2048 * 4);          // 73728 f16

  k_chansum<<<B_ * C_, 256, 0, stream>>>(x, sums);
  k_select <<<B_, 128, 0, stream>>>(sums, sel);
  k_wconv  <<<(COUT_ * KTOT + 255) / 256, 256, 0, stream>>>(w, whf);
  k_conv   <<<B_ * H_ * 2, 256, 0, stream>>>(x, whf, bias, sel, out);
}
